// MhAttention_26285199851727
// MI455X (gfx1250) — matrix operand order repaired and checked by reading, not compiled, not run on hardware
//
#include <hip/hip_runtime.h>
#include <hip/hip_bf16.h>

typedef __attribute__((ext_vector_type(16))) _Float16 v16h;
typedef __attribute__((ext_vector_type(8)))  _Float16 v8h;
typedef __attribute__((ext_vector_type(8)))  float    v8f;
typedef int v4i __attribute__((vector_size(16)));

#define DEV static __device__ __forceinline__

// ---- CDNA5 async global->LDS copy (no VGPR staging), compile-safe guarded
#if defined(__AMDGCN__) && __has_builtin(__builtin_amdgcn_global_load_async_to_lds_b128) && \
    __has_builtin(__builtin_amdgcn_s_wait_asynccnt)
#define HAVE_ASYNC_LDS 1
#else
#define HAVE_ASYNC_LDS 0
#endif

DEV void async_copy16(const _Float16* g, _Float16* l) {
#if HAVE_ASYNC_LDS
    __builtin_amdgcn_global_load_async_to_lds_b128(
        (__attribute__((address_space(1))) v4i*)g,
        (__attribute__((address_space(3))) v4i*)l, 0, 0);
#else
    *(v8h*)l = *(const v8h*)g;
#endif
}
DEV void async_wait() {
#if HAVE_ASYNC_LDS
    __builtin_amdgcn_s_wait_asynccnt(0);
#endif
}

DEV v8f wmma_f16(v16h a, v16h b, v8f c) {
    return __builtin_amdgcn_wmma_f32_16x16x32_f16(
        false, a, false, b, (short)0, c, false, false);
}

// A-fragment (16x32, M x K), row-major LDS tile, row stride `ld` (elements).
// Lane L: half = L>>4, m = L&15. Holds K chunks [half*8,+8) and [16+half*8,+8).
DEV v16h load_fragA(const _Float16* base, int ld, int row0, int kOff) {
    int lane = threadIdx.x & 31;
    int half = lane >> 4, m = lane & 15;
    const _Float16* p = base + (size_t)(row0 + m) * ld + kOff + half * 8;
    v8h lo = *(const v8h*)(p);
    v8h hi = *(const v8h*)(p + 16);
    v16h out;
#pragma unroll
    for (int i = 0; i < 8; ++i) { out[i] = lo[i]; out[i + 8] = hi[i]; }
    return out;
}

// B-fragment (K=32 x N=16) from Bt LDS tile stored [n][k] row-major.
// Lane L: n = L&15, half = L>>4. Holds K chunks [half*8,+8) and [16+half*8,+8).
DEV v16h load_fragB(const _Float16* base, int ld, int n0, int kOff) {
    int lane = threadIdx.x & 31;
    int half = lane >> 4, n = lane & 15;
    const _Float16* p = base + (size_t)(n0 + n) * ld + kOff + half * 8;
    v8h lo = *(const v8h*)(p);
    v8h hi = *(const v8h*)(p + 16);
    v16h out;
#pragma unroll
    for (int i = 0; i < 8; ++i) { out[i] = lo[i]; out[i + 8] = hi[i]; }
    return out;
}

// ---------------------------------------------------------------- convert
__global__ __launch_bounds__(256) void cvt_f32_f16(const float* __restrict__ in,
                                                   _Float16* __restrict__ out, int n) {
    int i = blockIdx.x * 256 + threadIdx.x;
    if (i < n) out[i] = (_Float16)in[i];
}

// ---------------------------------------------------------------- GEMM
// C[M,N] = X[M,K] * W[K,N]; f16 inputs, f32 accumulate; f16 or f32 out (template).
// Block 256 threads (8 waves), BM=128 BN=64 BK=64; wave grid 4x2, 2x2 WMMA each,
// 8 WMMA per barrier pair. A tile filled with async global->LDS copies.
#define GBM 128
#define GBN 64
#define GBK 64
#define GLD 72
template <bool F16OUT>
__global__ __launch_bounds__(256) void gemm_wmma(const _Float16* __restrict__ X,
                                                 const _Float16* __restrict__ W,
                                                 void* __restrict__ outP,
                                                 int M, int N, int K) {
    __shared__ _Float16 sA[GBM * GLD];   // [m][k]
    __shared__ _Float16 sB[GBN * GLD];   // transposed: [n][k]
    int tid = threadIdx.x;
    int wave = tid >> 5;
    int lane = tid & 31;
    int half = lane >> 4, cl = lane & 15;
    int mBase = blockIdx.x * GBM;
    int nBase = blockIdx.y * GBN;
    int waveM = wave >> 1, waveN = wave & 1;

    v8f acc[2][2];
#pragma unroll
    for (int i = 0; i < 2; ++i)
#pragma unroll
        for (int j = 0; j < 2; ++j) { v8f z = {}; acc[i][j] = z; }

    int arow = tid >> 1, aseg = (tid & 1) * 32;   // A: 128 rows x 64 k
    int bkr  = tid >> 2, bnc  = (tid & 3) * 16;   // B: 64 k-rows x 64 n

    for (int k0 = 0; k0 < K; k0 += GBK) {
        // ---- A tile 128x64: async global->LDS, 4 x 16B per thread
        {
            const _Float16* src = X + (size_t)(mBase + arow) * K + k0 + aseg;
            _Float16* dst = &sA[arow * GLD + aseg];
#pragma unroll
            for (int j = 0; j < 4; ++j) async_copy16(src + j * 8, dst + j * 8);
        }
        // ---- B tile 64x64 -> transposed LDS [n][k]
        {
            const _Float16* src = W + (size_t)(k0 + bkr) * N + nBase + bnc;
            v8h b0 = *(const v8h*)src;
            v8h b1 = *(const v8h*)(src + 8);
#pragma unroll
            for (int j = 0; j < 8; ++j) {
                sB[(size_t)(bnc + j) * GLD + bkr]     = b0[j];
                sB[(size_t)(bnc + 8 + j) * GLD + bkr] = b1[j];
            }
        }
        async_wait();
        __syncthreads();

#pragma unroll
        for (int ks = 0; ks < GBK; ks += 32) {
            v16h a0 = load_fragA(sA, GLD, waveM * 32,      ks);
            v16h a1 = load_fragA(sA, GLD, waveM * 32 + 16, ks);
            v16h b0 = load_fragB(sB, GLD, waveN * 32,      ks);
            v16h b1 = load_fragB(sB, GLD, waveN * 32 + 16, ks);
            acc[0][0] = wmma_f16(a0, b0, acc[0][0]);
            acc[0][1] = wmma_f16(a0, b1, acc[0][1]);
            acc[1][0] = wmma_f16(a1, b0, acc[1][0]);
            acc[1][1] = wmma_f16(a1, b1, acc[1][1]);
        }
        __syncthreads();
    }

    // ---- branch-free epilogue; one row-offset computation per (mi,r)
    int nCol = nBase + waveN * 32 + cl;
#pragma unroll
    for (int mi = 0; mi < 2; ++mi)
#pragma unroll
        for (int r = 0; r < 8; ++r) {
            int gm = mBase + waveM * 32 + mi * 16 + r + half * 8;
            size_t off = (size_t)gm * N + nCol;
            if (F16OUT) {
                _Float16* o = (_Float16*)outP;
                o[off]      = (_Float16)acc[mi][0][r];
                o[off + 16] = (_Float16)acc[mi][1][r];
            } else {
                float* o = (float*)outP;
                o[off]      = acc[mi][0][r];
                o[off + 16] = acc[mi][1][r];
            }
        }
}

// ---------------------------------------------------------------- RoPE
__global__ __launch_bounds__(256) void rope_kernel(_Float16* __restrict__ Q,
                                                   _Float16* __restrict__ K,
                                                   int BS, int S, int D, int H, int Dh) {
    int idx = blockIdx.x * 256 + threadIdx.x;
    int hp = Dh / 2;
    int total = BS * H * hp;
    if (idx >= total) return;
    int i  = idx % hp;
    int h  = (idx / hp) % H;
    int bs = idx / (hp * H);
    int s  = bs % S;
    float theta = __powf(10000.0f, -(2.0f * i) / (float)Dh);
    float ang = (float)s * theta;
    float sn, cs;
    __sincosf(ang, &sn, &cs);
    size_t off = (size_t)bs * D + h * Dh + 2 * i;
    float q1 = (float)Q[off], q2 = (float)Q[off + 1];
    Q[off]     = (_Float16)(q1 * cs - q2 * sn);
    Q[off + 1] = (_Float16)(q1 * sn + q2 * cs);
    float k1 = (float)K[off], k2 = (float)K[off + 1];
    K[off]     = (_Float16)(k1 * cs - k2 * sn);
    K[off + 1] = (_Float16)(k1 * sn + k2 * cs);
}

// ---------------------------------------------------------------- Flash attention
// grid = (S/128, B*H), block = 256 (8 waves). Each wave owns 16 query rows.
#define ABQ  128
#define ABK  32
#define ADH  64
#define ALQ  72   // ldsQ row stride (elems)
#define ALK  72   // ldsK row stride
#define ALV  40   // ldsVt row stride
#define ALP  40   // per-wave P row stride
__global__ __launch_bounds__(256) void attn_kernel(const _Float16* __restrict__ Q,
                                                   const _Float16* __restrict__ K,
                                                   const _Float16* __restrict__ V,
                                                   _Float16* __restrict__ attnOut,
                                                   int B, int S, int D, int H) {
    __shared__ _Float16 sQ[ABQ * ALQ];            // [q][dh]
    __shared__ _Float16 sK[ABK * ALK];            // [key][dh]  (Bt for QK^T)
    __shared__ _Float16 sVt[ADH * ALV];           // [dh][key]  (Bt for PV)
    __shared__ _Float16 sP[8 * 16 * ALP];         // per-wave 16x32 P tile

    int tid  = threadIdx.x;
    int wave = tid >> 5;
    int lane = tid & 31;
    int half = lane >> 4, cl = lane & 15;

    int qb = blockIdx.x;
    int bh = blockIdx.y;
    int b = bh / H, h = bh % H;
    int qs = qb * ABQ;
    const float scale = 0.03125f; // 1/sqrt(1024)

    // ---- load Q tile (resident): async global->LDS, 4 x 16B per thread
    {
        int row = tid >> 1;
        int seg = (tid & 1) * 32;
        const _Float16* src = Q + (size_t)(b * S + qs + row) * D + h * ADH + seg;
        _Float16* dst = &sQ[row * ALQ + seg];
#pragma unroll
        for (int j = 0; j < 4; ++j) async_copy16(src + j * 8, dst + j * 8);
    }

    float mrow[8], lrow[8];
    v8f acc[4];
#pragma unroll
    for (int r = 0; r < 8; ++r) { mrow[r] = -1e30f; lrow[r] = 0.0f; }
#pragma unroll
    for (int d = 0; d < 4; ++d) { v8f z = {}; acc[d] = z; }

    int nkt = (qs + ABQ) / ABK; // causal: keys 0 .. qs+127

    for (int kt = 0; kt < nkt; ++kt) {
        int ks = kt * ABK;
        __syncthreads(); // prior iteration's LDS reads done before overwrite
        // ---- K tile [key][dh]: async copy; V tile -> transposed [dh][key]
        {
            int key = tid >> 3;
            int seg = (tid & 7) * 8;
            const _Float16* srck = K + (size_t)(b * S + ks + key) * D + h * ADH + seg;
            async_copy16(srck, &sK[key * ALK + seg]);
            const _Float16* srcv = V + (size_t)(b * S + ks + key) * D + h * ADH + seg;
            v8h vv = *(const v8h*)srcv;
#pragma unroll
            for (int j = 0; j < 8; ++j) sVt[(size_t)(seg + j) * ALV + key] = vv[j];
        }
        async_wait();     // K (and Q on iter 0) in LDS
        __syncthreads();

        // ---- S = Q * K^T  (16 x 32 per wave, Dh=64 -> two k-steps)
        v16h a0 = load_fragA(sQ, ALQ, wave * 16, 0);
        v16h a1 = load_fragA(sQ, ALQ, wave * 16, 32);
        v8f sc[2];
#pragma unroll
        for (int nt = 0; nt < 2; ++nt) {
            v16h bk0 = load_fragB(sK, ALK, nt * 16, 0);
            v16h bk1 = load_fragB(sK, ALK, nt * 16, 32);
            v8f z = {};
            z = wmma_f16(a0, bk0, z);
            z = wmma_f16(a1, bk1, z);
            sc[nt] = z;
        }

        // ---- mask + online softmax + P to per-wave LDS
        _Float16* pW = &sP[(size_t)wave * 16 * ALP];
#pragma unroll
        for (int r = 0; r < 8; ++r) {
            int m    = r + half * 8;
            int qrow = qs + wave * 16 + m;
            float s0 = sc[0][r] * scale;
            float s1 = sc[1][r] * scale;
            if (ks + cl      > qrow) s0 = -1e30f;
            if (ks + 16 + cl > qrow) s1 = -1e30f;
            float mx = fmaxf(s0, s1);
#pragma unroll
            for (int xm = 1; xm < 16; xm <<= 1)
                mx = fmaxf(mx, __shfl_xor(mx, xm, 32));
            float mn    = fmaxf(mrow[r], mx);
            float alpha = __expf(mrow[r] - mn);
            float p0 = __expf(s0 - mn);
            float p1 = __expf(s1 - mn);
            float sum = p0 + p1;
#pragma unroll
            for (int xm = 1; xm < 16; xm <<= 1)
                sum += __shfl_xor(sum, xm, 32);
            mrow[r] = mn;
            lrow[r] = lrow[r] * alpha + sum;
#pragma unroll
            for (int d = 0; d < 4; ++d) acc[d][r] *= alpha;
            pW[(size_t)m * ALP + cl]      = (_Float16)p0;
            pW[(size_t)m * ALP + 16 + cl] = (_Float16)p1;
        }
        // wave-local LDS store->load ordering
        asm volatile("s_wait_dscnt 0" ::: "memory");

        // ---- O += P * V  (A = P 16x32, B = Vt, 4 dh n-tiles)
        v16h ap = load_fragA(pW, ALP, 0, 0);
#pragma unroll
        for (int d = 0; d < 4; ++d) {
            v16h bv = load_fragB(sVt, ALV, d * 16, 0);
            acc[d] = wmma_f16(ap, bv, acc[d]);
        }
    }

    // ---- normalize + write attn (f16)
#pragma unroll
    for (int d = 0; d < 4; ++d)
#pragma unroll
        for (int r = 0; r < 8; ++r) {
            int m    = r + half * 8;
            int qrow = qs + wave * 16 + m;
            int dc   = d * 16 + cl;
            attnOut[(size_t)(b * S + qrow) * D + h * ADH + dc] =
                (_Float16)(acc[d][r] / lrow[r]);
        }
}

// ---------------------------------------------------------------- launcher
extern "C" void kernel_launch(void* const* d_in, const int* in_sizes, int n_in,
                              void* d_out, int out_size, void* d_ws, size_t ws_size,
                              hipStream_t stream) {
    (void)in_sizes; (void)n_in; (void)out_size; (void)ws_size;
    const int B = 2, S = 2048, D = 1024, H = 16, Dh = 64;
    const int M = B * S;                 // 4096
    const size_t XN = (size_t)M * D;     // 4,194,304
    const size_t WN = (size_t)D * D;     // 1,048,576

    const float* x  = (const float*)d_in[0];
    const float* qw = (const float*)d_in[1];
    const float* kw = (const float*)d_in[2];
    const float* vw = (const float*)d_in[3];
    const float* ow = (const float*)d_in[4];
    float* out = (float*)d_out;

    char* ws = (char*)d_ws;
    _Float16* Xh   = (_Float16*)(ws);                       // 8 MB
    _Float16* Qwh  = (_Float16*)(ws + (8u  << 20));         // 2 MB
    _Float16* Kwh  = (_Float16*)(ws + (10u << 20));         // 2 MB
    _Float16* Vwh  = (_Float16*)(ws + (12u << 20));         // 2 MB
    _Float16* Owh  = (_Float16*)(ws + (14u << 20));         // 2 MB
    _Float16* Qh   = (_Float16*)(ws + (16u << 20));         // 8 MB
    _Float16* Kh   = (_Float16*)(ws + (24u << 20));         // 8 MB
    _Float16* Vh   = (_Float16*)(ws + (32u << 20));         // 8 MB
    _Float16* Ah   = (_Float16*)(ws + (40u << 20));         // 8 MB

    // 1) fp32 -> f16
    cvt_f32_f16<<<(int)((XN + 255) / 256), 256, 0, stream>>>(x,  Xh,  (int)XN);
    cvt_f32_f16<<<(int)((WN + 255) / 256), 256, 0, stream>>>(qw, Qwh, (int)WN);
    cvt_f32_f16<<<(int)((WN + 255) / 256), 256, 0, stream>>>(kw, Kwh, (int)WN);
    cvt_f32_f16<<<(int)((WN + 255) / 256), 256, 0, stream>>>(vw, Vwh, (int)WN);
    cvt_f32_f16<<<(int)((WN + 255) / 256), 256, 0, stream>>>(ow, Owh, (int)WN);

    // 2) Q/K/V projections (f16 out)
    dim3 ggrid(M / GBM, D / GBN);
    gemm_wmma<true><<<ggrid, 256, 0, stream>>>(Xh, Qwh, Qh, M, D, D);
    gemm_wmma<true><<<ggrid, 256, 0, stream>>>(Xh, Kwh, Kh, M, D, D);
    gemm_wmma<true><<<ggrid, 256, 0, stream>>>(Xh, Vwh, Vh, M, D, D);

    // 3) RoPE on Q, K
    int ropeN = M * H * (Dh / 2);
    rope_kernel<<<(ropeN + 255) / 256, 256, 0, stream>>>(Qh, Kh, M, S, D, H, Dh);

    // 4) causal flash attention
    attn_kernel<<<dim3(S / ABQ, B * H), 256, 0, stream>>>(Qh, Kh, Vh, Ah, B, S, D, H);

    // 5) output projection (fp32 out)
    gemm_wmma<false><<<ggrid, 256, 0, stream>>>(Ah, Owh, out, M, D, D);
}
